// Causal3DTransformerBlock_36842229465889
// MI455X (gfx1250) — compile-verified, matrix operand order checked by reading, not run on hardware
//
#include <hip/hip_runtime.h>
#include <hip/hip_bf16.h>
#include <math.h>

typedef __bf16 bf16_t;
typedef __attribute__((ext_vector_type(16))) __bf16 v16bf;
typedef __attribute__((ext_vector_type(8)))  float  v8f;

#define WMMA_BF16(A,B,C) __builtin_amdgcn_wmma_f32_16x16x32_bf16(false,(A),false,(B),(short)0,(C),false,false)

// async copy global -> LDS (CDNA5): vdst = 32-bit LDS byte offset, vaddr = 64-bit global addr
#define ASYNC_LD_B128(ldsoff, gptr) \
    asm volatile("global_load_async_to_lds_b128 %0, %1, off" :: "v"(ldsoff), "v"(gptr) : "memory")
#define WAIT_ASYNC0() \
    asm volatile("s_wait_asynccnt 0x0" ::: "memory")

#define Bsz 2
#define Ssz 2048
#define Csz 768
#define Hsz 8
#define Dh  96
#define CTX 128
#define Isz 3072
#define Msz (Bsz*Ssz)   // 4096

// ---------------- weight transpose + bf16 convert: W[K,N] f32 -> WT[N,K] bf16
__global__ void wtrans_kernel(const float* __restrict__ W, bf16_t* __restrict__ WT,
                              int K_, int N_) {
    size_t i = (size_t)blockIdx.x * blockDim.x + threadIdx.x;
    if (i >= (size_t)K_ * N_) return;
    int k = (int)(i / N_);
    int n = (int)(i % N_);
    WT[(size_t)n * K_ + k] = (bf16_t)W[i];
}

// ---------------- plain f32 -> bf16 convert
__global__ void cvt_kernel(const float* __restrict__ s, bf16_t* __restrict__ d, int n) {
    int i = blockIdx.x * blockDim.x + threadIdx.x;
    if (i < n) d[i] = (bf16_t)s[i];
}

// ---------------- LayerNorm: one row per block, bf16 output
__global__ void ln_kernel(const float* __restrict__ x, const float* __restrict__ g,
                          const float* __restrict__ bta, bf16_t* __restrict__ out) {
    int row = blockIdx.x;
    const float* xr = x + (size_t)row * Csz;
    __shared__ float red[256];
    float s = 0.f;
    for (int i = threadIdx.x; i < Csz; i += 256) s += xr[i];
    red[threadIdx.x] = s; __syncthreads();
    for (int o = 128; o > 0; o >>= 1) { if (threadIdx.x < o) red[threadIdx.x] += red[threadIdx.x + o]; __syncthreads(); }
    float mu = red[0] / Csz;
    __syncthreads();
    float s2 = 0.f;
    for (int i = threadIdx.x; i < Csz; i += 256) { float d = xr[i] - mu; s2 += d * d; }
    red[threadIdx.x] = s2; __syncthreads();
    for (int o = 128; o > 0; o >>= 1) { if (threadIdx.x < o) red[threadIdx.x] += red[threadIdx.x + o]; __syncthreads(); }
    float inv = rsqrtf(red[0] / Csz + 1e-5f);
    bf16_t* orow = out + (size_t)row * Csz;
    for (int i = threadIdx.x; i < Csz; i += 256)
        orow[i] = (bf16_t)((xr[i] - mu) * inv * g[i] + bta[i]);
}

// ---------------- LDS-staged, async double-buffered bf16 WMMA GEMM
// out[M,N] = A[M,K] @ BT[N,K]^T. Block tile 128x128, 8 waves (4x2), each wave 32x64
// (8 WMMAs per 32-K step). A/B K-slabs staged into LDS once per block via
// global_load_async_to_lds_b128 (ASYNCcnt), double-buffered.
// mode 0: outF = acc ; mode 1: outF = acc + res ; mode 2: outB = silu(gate)*acc (bf16)
__global__ void gemm_bf16_kernel(const bf16_t* __restrict__ A, const bf16_t* __restrict__ BT,
                                 int M, int N, int K,
                                 float* __restrict__ outF, const float* __restrict__ res,
                                 const float* __restrict__ gate, bf16_t* __restrict__ outB,
                                 int mode) {
    __shared__ __align__(16) bf16_t lA[2][128 * 32];   // 2 x 8KB
    __shared__ __align__(16) bf16_t lB[2][128 * 32];   // 2 x 8KB
    int tid  = threadIdx.x;
    int lane = tid & 31, wid = tid >> 5;
    int wm = wid >> 1, wn = wid & 1;                   // 4x2 wave grid
    int mB = blockIdx.y * 128, nB = blockIdx.x * 128;
    int lm = lane & 15, hf = lane >> 4;

    // loader mapping: thread -> (row 0..63, 16B segment 0..3); each thread stages
    // 2 rows of A and 2 rows of B (rows r and r+64) per K-slab.
    int lrow = tid >> 2;
    int seg  = (tid & 3) * 8;                          // bf16 element offset (16B units)
    const bf16_t* gA = A  + (size_t)(mB + lrow) * K + seg;
    const bf16_t* gB = BT + (size_t)(nB + lrow) * K + seg;
    size_t rowskip = (size_t)64 * K;                   // +64 rows in global
    unsigned ldsRow = (unsigned)(lrow * 32 + seg) * 2; // byte offset within a buffer

    unsigned baseA0 = (unsigned)(size_t)&lA[0][0];
    unsigned baseB0 = (unsigned)(size_t)&lB[0][0];

    auto stage = [&](int kk, int buf) {
        unsigned oa = baseA0 + (unsigned)buf * 8192 + ldsRow;
        unsigned ob = baseB0 + (unsigned)buf * 8192 + ldsRow;
        ASYNC_LD_B128(oa,        gA + kk);
        ASYNC_LD_B128(oa + 4096, gA + rowskip + kk);
        ASYNC_LD_B128(ob,        gB + kk);
        ASYNC_LD_B128(ob + 4096, gB + rowskip + kk);
    };

    v8f acc[2][4];
    for (int i = 0; i < 2; i++)
        for (int j = 0; j < 4; j++) acc[i][j] = (v8f){};

    int nk = K >> 5;
    stage(0, 0);
    for (int kt = 0; kt < nk; kt++) {
        int buf = kt & 1;
        WAIT_ASYNC0();            // my async stores to LDS done
        __syncthreads();          // everyone's staged data visible
        if (kt + 1 < nk) stage((kt + 1) << 5, buf ^ 1);
        const bf16_t* sA = &lA[buf][0];
        const bf16_t* sB = &lB[buf][0];
        v16bf a0 = *(const v16bf*)(sA + (wm * 32 + lm) * 32 + hf * 16);
        v16bf a1 = *(const v16bf*)(sA + (wm * 32 + 16 + lm) * 32 + hf * 16);
        #pragma unroll
        for (int j = 0; j < 4; j++) {
            v16bf b = *(const v16bf*)(sB + (wn * 64 + j * 16 + lm) * 32 + hf * 16);
            acc[0][j] = WMMA_BF16(a0, b, acc[0][j]);
            acc[1][j] = WMMA_BF16(a1, b, acc[1][j]);
        }
        __syncthreads();          // reads of buf done before it is restaged
    }

    int m0 = mB + wm * 32, n0 = nB + wn * 64;
    for (int i = 0; i < 2; i++)
        for (int j = 0; j < 4; j++)
            for (int v = 0; v < 8; v++) {
                int m = m0 + i * 16 + v + hf * 8;
                int n = n0 + j * 16 + lm;
                size_t idx = (size_t)m * N + n;
                float val = acc[i][j][v];
                if (mode == 1) {
                    outF[idx] = val + res[idx];
                } else if (mode == 2) {
                    float gg = gate[idx];
                    float sg = gg / (1.f + __expf(-gg));
                    outB[idx] = (bf16_t)(sg * val);
                } else {
                    outF[idx] = val;
                }
            }
}

// ---------------- RoPE-3D + head scatter: qkv f32 [B,S,3C] -> qh/kh bf16 [B,H,S,Dh], vT bf16 [B,H,Dh,S]
__global__ void rope_scatter_kernel(const float* __restrict__ qkv,
                                    bf16_t* __restrict__ qh, bf16_t* __restrict__ kh,
                                    bf16_t* __restrict__ vt) {
    int s = blockIdx.x;          // 0..2047
    int b = blockIdx.y;          // 0..1
    int d = threadIdx.x;         // 0..95
    int xw = s & 15, y = (s >> 4) & 15, t = s >> 8;
    float pos; int dm;
    if      (d < 32) { pos = (float)xw; dm = d & 15; }
    else if (d < 64) { pos = (float)y;  dm = (d - 32) & 15; }
    else             { pos = (float)t;  dm = (d - 64) & 15; }
    float inv = __powf(10000.f, -(float)(2 * dm) / 32.f);
    float ang = pos * inv;
    float c = __cosf(ang), sn = __sinf(ang);
    float sgn = (d < 48) ? -1.f : 1.f;
    int dp = (d < 48) ? d + 48 : d - 48;
    const float* base = qkv + ((size_t)b * Ssz + s) * (3 * Csz);
    for (int h = 0; h < Hsz; h++) {
        float qv = base[h * Dh + d];
        float kv = base[Csz + h * Dh + d];
        float vv = base[2 * Csz + h * Dh + d];
        float qp = base[h * Dh + dp];
        float kp = base[Csz + h * Dh + dp];
        float qo = qv * c + sgn * qp * sn;
        float ko = kv * c + sgn * kp * sn;
        size_t o1 = ((size_t)(b * Hsz + h) * Ssz + s) * Dh + d;
        qh[o1] = (bf16_t)qo;
        kh[o1] = (bf16_t)ko;
        vt[((size_t)(b * Hsz + h) * Dh + d) * Ssz + s] = (bf16_t)vv;
    }
}

// ---------------- cross-attn head scatter: src f32 [B,T,C] -> dst bf16 [B,H,T,Dh] (or [B,H,Dh,T] if transposed)
__global__ void scatter_heads_kernel(const float* __restrict__ src, bf16_t* __restrict__ dst,
                                     int T, int transposed) {
    size_t i = (size_t)blockIdx.x * blockDim.x + threadIdx.x;
    size_t total = (size_t)Bsz * T * Csz;
    if (i >= total) return;
    int cidx = (int)(i % Csz);
    size_t bt = i / Csz;
    int tt = (int)(bt % T);
    int b  = (int)(bt / T);
    int h = cidx / Dh, d = cidx % Dh;
    float v = src[i];
    if (transposed) dst[((size_t)(b * Hsz + h) * Dh + d) * T + tt] = (bf16_t)v;
    else            dst[((size_t)(b * Hsz + h) * T + tt) * Dh + d] = (bf16_t)v;
}

__device__ __forceinline__ float half_max(float v) {
    for (int m = 1; m < 16; m <<= 1) v = fmaxf(v, __shfl_xor(v, m, 32));
    return v;
}
__device__ __forceinline__ float half_sum(float v) {
    for (int m = 1; m < 16; m <<= 1) v += __shfl_xor(v, m, 32);
    return v;
}

// ---------------- flash attention: 4 waves per block, each wave owns 16 query rows of one (b,h)
// Q [B,H,S,Dh] bf16, Kb [B,H,T,Dh] bf16, VT [B,H,Dh,T] bf16, O [B,S,C] bf16
__global__ void attn_kernel(const bf16_t* __restrict__ Q, const bf16_t* __restrict__ Kb,
                            const bf16_t* __restrict__ VT, bf16_t* __restrict__ O,
                            int S_, int T, int causal, float scale) {
    __shared__ float lds[4 * 16 * 32];               // per-wave 2KB slab
    int lane = threadIdx.x & 31, wid = threadIdx.x >> 5;
    float* slab = lds + wid * 512;
    int lm = lane & 15, hf = lane >> 4, h16 = hf * 16;
    int qt = blockIdx.x * 4 + wid;
    int h = blockIdx.y, b = blockIdx.z;

    const bf16_t* qbase = Q + ((size_t)(b * Hsz + h) * S_ + qt * 16 + lm) * Dh + h16;
    v16bf aq0 = *(const v16bf*)(qbase);
    v16bf aq1 = *(const v16bf*)(qbase + 32);
    v16bf aq2 = *(const v16bf*)(qbase + 64);

    v8f acc[6];
    for (int t = 0; t < 6; t++) acc[t] = (v8f){};
    float mrow[8], lrow[8];
    for (int v = 0; v < 8; v++) { mrow[v] = -__builtin_inff(); lrow[v] = 0.f; }

    int nkt = causal ? (qt * 16 + 16 + 31) / 32 : (T + 31) / 32;
    const bf16_t* kbb = Kb + (size_t)(b * Hsz + h) * T * Dh;
    const bf16_t* vbb = VT + (size_t)(b * Hsz + h) * Dh * T;

    for (int kt = 0; kt < nkt; kt++) {
        int k0 = kt * 32;
        v8f s0 = {}, s1 = {};
        const bf16_t* kr0 = kbb + (size_t)(k0 + lm) * Dh + h16;
        const bf16_t* kr1 = kbb + (size_t)(k0 + 16 + lm) * Dh + h16;
        {
            v16bf b0 = *(const v16bf*)(kr0);
            v16bf b1 = *(const v16bf*)(kr1);
            s0 = WMMA_BF16(aq0, b0, s0); s1 = WMMA_BF16(aq0, b1, s1);
            b0 = *(const v16bf*)(kr0 + 32); b1 = *(const v16bf*)(kr1 + 32);
            s0 = WMMA_BF16(aq1, b0, s0); s1 = WMMA_BF16(aq1, b1, s1);
            b0 = *(const v16bf*)(kr0 + 64); b1 = *(const v16bf*)(kr1 + 64);
            s0 = WMMA_BF16(aq2, b0, s0); s1 = WMMA_BF16(aq2, b1, s1);
        }
        float p0[8], p1[8];
        for (int v = 0; v < 8; v++) {
            float a0 = s0[v] * scale, a1 = s1[v] * scale;
            if (causal) {
                int qrow = qt * 16 + v + hf * 8;
                if (k0 + lm > qrow)      a0 = -__builtin_inff();
                if (k0 + 16 + lm > qrow) a1 = -__builtin_inff();
            }
            float rmax = half_max(fmaxf(a0, a1));
            float mnew = fmaxf(mrow[v], rmax);
            float corr = __expf(mrow[v] - mnew);
            float e0 = __expf(a0 - mnew), e1 = __expf(a1 - mnew);
            float rs = half_sum(e0 + e1);
            lrow[v] = lrow[v] * corr + rs;
            mrow[v] = mnew;
            p0[v] = e0; p1[v] = e1;
            for (int t = 0; t < 6; t++) acc[t][v] *= corr;
        }
        // repack P from C-layout into A-fragment layout via per-wave LDS slab (DS in-order per wave)
        for (int v = 0; v < 8; v++) {
            int r = v + hf * 8;
            slab[r * 32 + lm]      = p0[v];
            slab[r * 32 + 16 + lm] = p1[v];
        }
        v16bf ap;
        for (int j = 0; j < 16; j++) ap[j] = (bf16_t)slab[lm * 32 + h16 + j];
        for (int nt = 0; nt < 6; nt++) {
            const bf16_t* vr = vbb + (size_t)(nt * 16 + lm) * T + k0 + h16;
            v16bf bv = *(const v16bf*)(vr);
            acc[nt] = WMMA_BF16(ap, bv, acc[nt]);
        }
    }
    for (int nt = 0; nt < 6; nt++)
        for (int v = 0; v < 8; v++) {
            int q = qt * 16 + v + hf * 8;
            O[((size_t)b * S_ + q) * Csz + h * Dh + nt * 16 + lm] = (bf16_t)(acc[nt][v] / lrow[v]);
        }
}

extern "C" void kernel_launch(void* const* d_in, const int* in_sizes, int n_in,
                              void* d_out, int out_size, void* d_ws, size_t ws_size,
                              hipStream_t stream) {
    const float* x        = (const float*)d_in[0];
    const float* context  = (const float*)d_in[1];
    const float* wqkv     = (const float*)d_in[2];
    const float* w_attn   = (const float*)d_in[3];
    const float* ln1_g    = (const float*)d_in[4];
    const float* ln1_b    = (const float*)d_in[5];
    const float* wq_c     = (const float*)d_in[6];
    const float* wk_c     = (const float*)d_in[7];
    const float* wv_c     = (const float*)d_in[8];
    const float* w_cross  = (const float*)d_in[9];
    const float* ln2_g    = (const float*)d_in[10];
    const float* ln2_b    = (const float*)d_in[11];
    const float* w_gate   = (const float*)d_in[12];
    const float* w_up     = (const float*)d_in[13];
    const float* w_down   = (const float*)d_in[14];
    const float* ln3_g    = (const float*)d_in[15];
    const float* ln3_b    = (const float*)d_in[16];

    char* ws = (char*)d_ws;
    size_t off = 0;
    auto alloc = [&](size_t bytes) -> char* {
        char* p = ws + off;
        off = (off + bytes + 255) & ~(size_t)255;
        return p;
    };

    bf16_t* wqkvT  = (bf16_t*)alloc((size_t)2304 * 768 * 2);
    bf16_t* wattnT = (bf16_t*)alloc((size_t)768 * 768 * 2);
    bf16_t* wqcT   = (bf16_t*)alloc((size_t)768 * 768 * 2);
    bf16_t* wkcT   = (bf16_t*)alloc((size_t)768 * 768 * 2);
    bf16_t* wvcT   = (bf16_t*)alloc((size_t)768 * 768 * 2);
    bf16_t* wcrsT  = (bf16_t*)alloc((size_t)768 * 768 * 2);
    bf16_t* wgateT = (bf16_t*)alloc((size_t)3072 * 768 * 2);
    bf16_t* wupT   = (bf16_t*)alloc((size_t)3072 * 768 * 2);
    bf16_t* wdownT = (bf16_t*)alloc((size_t)768 * 3072 * 2);
    bf16_t* xn     = (bf16_t*)alloc((size_t)Msz * Csz * 2);
    bf16_t* ctxb   = (bf16_t*)alloc((size_t)Bsz * CTX * Csz * 2);
    float*  qkv    = (float*) alloc((size_t)Msz * 3 * Csz * 4);
    bf16_t* qh     = (bf16_t*)alloc((size_t)Bsz * Hsz * Ssz * Dh * 2);
    bf16_t* kh     = (bf16_t*)alloc((size_t)Bsz * Hsz * Ssz * Dh * 2);
    bf16_t* vt     = (bf16_t*)alloc((size_t)Bsz * Hsz * Ssz * Dh * 2);
    bf16_t* ob     = (bf16_t*)alloc((size_t)Msz * Csz * 2);
    float*  x1     = (float*) alloc((size_t)Msz * Csz * 4);
    float*  x2     = (float*) alloc((size_t)Msz * Csz * 4);
    float*  qc     = (float*) alloc((size_t)Msz * Csz * 4);
    float*  kc     = (float*) alloc((size_t)Bsz * CTX * Csz * 4);
    float*  vc     = (float*) alloc((size_t)Bsz * CTX * Csz * 4);
    bf16_t* khc    = (bf16_t*)alloc((size_t)Bsz * Hsz * CTX * Dh * 2);
    bf16_t* vtc    = (bf16_t*)alloc((size_t)Bsz * Hsz * CTX * Dh * 2);
    float*  gbuf   = (float*) alloc((size_t)Msz * Isz * 4);
    bf16_t* gu     = (bf16_t*)alloc((size_t)Msz * Isz * 2);

    float scale = 1.0f / sqrtf((float)Dh);

    auto wt = [&](const float* W, bf16_t* WT, int K_, int N_) {
        size_t n = (size_t)K_ * N_;
        wtrans_kernel<<<(unsigned)((n + 255) / 256), 256, 0, stream>>>(W, WT, K_, N_);
    };
    wt(wqkv,    wqkvT,  768, 2304);
    wt(w_attn,  wattnT, 768, 768);
    wt(wq_c,    wqcT,   768, 768);
    wt(wk_c,    wkcT,   768, 768);
    wt(wv_c,    wvcT,   768, 768);
    wt(w_cross, wcrsT,  768, 768);
    wt(w_gate,  wgateT, 768, 3072);
    wt(w_up,    wupT,   768, 3072);
    wt(w_down,  wdownT, 3072, 768);

    int nctx = Bsz * CTX * Csz;
    cvt_kernel<<<(nctx + 255) / 256, 256, 0, stream>>>(context, ctxb, nctx);

    // ---- self attention ----
    ln_kernel<<<Msz, 256, 0, stream>>>(x, ln1_g, ln1_b, xn);
    gemm_bf16_kernel<<<dim3(2304 / 128, Msz / 128), 256, 0, stream>>>(
        xn, wqkvT, Msz, 2304, 768, qkv, nullptr, nullptr, nullptr, 0);
    rope_scatter_kernel<<<dim3(Ssz, Bsz), Dh, 0, stream>>>(qkv, qh, kh, vt);
    attn_kernel<<<dim3(Ssz / 64, Hsz, Bsz), 128, 0, stream>>>(
        qh, kh, vt, ob, Ssz, Ssz, 1, scale);
    gemm_bf16_kernel<<<dim3(768 / 128, Msz / 128), 256, 0, stream>>>(
        ob, wattnT, Msz, 768, 768, x1, x, nullptr, nullptr, 1);

    // ---- cross attention ----
    ln_kernel<<<Msz, 256, 0, stream>>>(x1, ln2_g, ln2_b, xn);
    gemm_bf16_kernel<<<dim3(768 / 128, Msz / 128), 256, 0, stream>>>(
        xn, wqcT, Msz, 768, 768, qc, nullptr, nullptr, nullptr, 0);
    gemm_bf16_kernel<<<dim3(768 / 128, (Bsz * CTX) / 128), 256, 0, stream>>>(
        ctxb, wkcT, Bsz * CTX, 768, 768, kc, nullptr, nullptr, nullptr, 0);
    gemm_bf16_kernel<<<dim3(768 / 128, (Bsz * CTX) / 128), 256, 0, stream>>>(
        ctxb, wvcT, Bsz * CTX, 768, 768, vc, nullptr, nullptr, nullptr, 0);
    {
        size_t nq = (size_t)Msz * Csz;
        scatter_heads_kernel<<<(unsigned)((nq + 255) / 256), 256, 0, stream>>>(qc, qh, Ssz, 0);
        size_t nk = (size_t)Bsz * CTX * Csz;
        scatter_heads_kernel<<<(unsigned)((nk + 255) / 256), 256, 0, stream>>>(kc, khc, CTX, 0);
        scatter_heads_kernel<<<(unsigned)((nk + 255) / 256), 256, 0, stream>>>(vc, vtc, CTX, 1);
    }
    attn_kernel<<<dim3(Ssz / 64, Hsz, Bsz), 128, 0, stream>>>(
        qh, khc, vtc, ob, Ssz, CTX, 0, scale);
    gemm_bf16_kernel<<<dim3(768 / 128, Msz / 128), 256, 0, stream>>>(
        ob, wcrsT, Msz, 768, 768, x2, x1, nullptr, nullptr, 1);

    // ---- MLP ----
    ln_kernel<<<Msz, 256, 0, stream>>>(x2, ln3_g, ln3_b, xn);
    gemm_bf16_kernel<<<dim3(Isz / 128, Msz / 128), 256, 0, stream>>>(
        xn, wgateT, Msz, Isz, 768, gbuf, nullptr, nullptr, nullptr, 0);
    gemm_bf16_kernel<<<dim3(Isz / 128, Msz / 128), 256, 0, stream>>>(
        xn, wupT, Msz, Isz, 768, nullptr, nullptr, gbuf, gu, 2);
    gemm_bf16_kernel<<<dim3(768 / 128, Msz / 128), 256, 0, stream>>>(
        gu, wdownT, Msz, 768, 3072, (float*)d_out, x2, nullptr, nullptr, 1);
}
